// GraphConv_10703058501941
// MI455X (gfx1250) — compile-verified, matrix operand order checked by reading, not run on hardware
//
#include <hip/hip_runtime.h>
#include <hip/hip_bf16.h>

typedef __attribute__((ext_vector_type(16))) __bf16 v16bf;
typedef __attribute__((ext_vector_type(8)))  __bf16 v8bf;
typedef __attribute__((ext_vector_type(2)))  __bf16 v2bf;
typedef __attribute__((ext_vector_type(8)))  float  v8f;

#define C_IN   64
#define C_OUT  64
#define KDIM   128          // 2*C_IN
#define WAVES  8            // waves per block (256 threads, wave32)

__device__ __forceinline__ unsigned int pack2bf(float lo, float hi) {
#if __has_builtin(__builtin_amdgcn_cvt_pk_bf16_f32)
    union { v2bf v; unsigned int u; } cv;
    cv.v = __builtin_amdgcn_cvt_pk_bf16_f32(lo, hi);   // 1 VALU op, RNE
    return cv.u;
#else
    unsigned int a = __float_as_uint(lo), b = __float_as_uint(hi);
    a = (a + 0x7FFFu + ((a >> 16) & 1u)) >> 16;
    b = (b + 0x7FFFu + ((b >> 16) & 1u)) >> 16;
    return (b << 16) | (a & 0xFFFFu);
#endif
}

__device__ __forceinline__ unsigned short f2bf1(float f) {
    unsigned int u = __float_as_uint(f);
    return (unsigned short)((u + 0x7FFFu + ((u >> 16) & 1u)) >> 16);
}

// -------------------------------------------------------------------------
// Zero-init output (max-identity: messages are >= 0 after relu, and empty
// nodes must end at 0, matching the reference's where(isfinite, ..., 0)).
__global__ void ec_zero(float* __restrict__ out, int n) {
    int i = blockIdx.x * blockDim.x + threadIdx.x;
    if (i < n) out[i] = 0.0f;
}

// -------------------------------------------------------------------------
// Pack W [128 x 64] f32 row-major into bf16 B-fragments for
// v_wmma_f32_16x16x32_bf16. Fragment (t = N-tile 0..3, c = K-chunk 0..3):
//   lane n<16  -> col 16t+n,    elems 0..7 = K 32c+0..7,  8..15 = K 32c+16..23
//   lane n>=16 -> col 16t+n-16, elems 0..7 = K 32c+8..15, 8..15 = K 32c+24..31
__global__ void ec_pack_w(const float* __restrict__ W, unsigned short* __restrict__ wpk) {
    int tid = blockIdx.x * blockDim.x + threadIdx.x;
    if (tid >= 4 * 4 * 32) return;
    int lane = tid & 31;
    int c    = (tid >> 5) & 3;
    int t    = tid >> 7;
    int col  = t * 16 + (lane & 15);
    int hi   = (lane >> 4) << 3;                // 0 or 8
    unsigned short* dstp = wpk + (size_t)tid * 16;
#pragma unroll
    for (int j = 0; j < 16; ++j) {
        int k = 32 * c + hi + j + ((j >= 8) ? 8 : 0);
        dstp[j] = f2bf1(W[k * C_OUT + col]);
    }
}

// -------------------------------------------------------------------------
// Persistent main kernel: each wave loops over tiles of 16 edges.
// B fragments + bias live in registers for the whole kernel; waves run
// fully decoupled (per-wave LDS region + s_wait_dscnt, no block barrier).
__global__ void __launch_bounds__(WAVES * 32)
ec_edgeconv_wmma(const float* __restrict__ x,
                 const int*   __restrict__ srcp,
                 const int*   __restrict__ dstp,
                 const unsigned short* __restrict__ wpk,
                 const float* __restrict__ bias,
                 float*       __restrict__ out,
                 int E, int nTiles, int nIter) {
    __shared__ __align__(16) __bf16 feat[WAVES][16][KDIM];   // 32 KB

    const int lane   = threadIdx.x & 31;
    const int wave   = threadIdx.x >> 5;
    const int waveId = blockIdx.x * WAVES + wave;
    const int stride = gridDim.x * WAVES;

    const int colL = lane & 15;
    const int half = lane >> 4;                 // 0: edges 0..7 / K lo, 1: 8..15 / K hi
    const int hi8  = half << 3;                 // 0 or 8

    // ---- hoisted: per-lane bias and all 16 B fragments (register resident)
    float bv[4];
#pragma unroll
    for (int t = 0; t < 4; ++t) bv[t] = bias[t * 16 + colL];

    v16bf bf[4][4];
#pragma unroll
    for (int t = 0; t < 4; ++t)
#pragma unroll
        for (int c = 0; c < 4; ++c)
            bf[t][c] = *(const v16bf*)(wpk + (size_t)(((t * 4 + c) * 32 + lane) * 16));

    unsigned int* fRow32 = (unsigned int*)&feat[wave][0][0];
    unsigned int* outU   = (unsigned int*)out;

    for (int it = 0; it < nIter; ++it) {
        int tile = waveId + it * stride;
        if (tile >= nTiles) tile = nTiles - 1;  // duplicates idempotent under max
        const int base = tile * 16;

        // lanes 0..15: dst[base+lane], lanes 16..31: src[base+lane-16]
        int eClamp = base + colL;
        if (eClamp >= E) eClamp = E - 1;
        const int myIdx = (lane < 16) ? dstp[eClamp] : srcp[eClamp];

        // ---- gather: build feat tile [16 x 128] bf16 in LDS.
        // Row index e is wave-uniform -> readlane gives SGPR node ids, so the
        // x-row loads are saddr-based and fully coalesced (32 lanes x float2).
#pragma unroll
        for (int e = 0; e < 16; ++e) {
            int d = __builtin_amdgcn_readlane(myIdx, e);        // dst node
            int s = __builtin_amdgcn_readlane(myIdx, e + 16);   // src node
            const float2* xd = (const float2*)(x + (size_t)d * C_IN);
            const float2* xs = (const float2*)(x + (size_t)s * C_IN);
            float2 xi = xd[lane];               // cols 2*lane, 2*lane+1
            float2 xj = xs[lane];
            fRow32[(e * KDIM) / 2 + lane]        = pack2bf(xi.x, xi.y);
            fRow32[(e * KDIM + C_IN) / 2 + lane] = pack2bf(xj.x - xi.x,
                                                           xj.y - xi.y);
        }
        // DS ops are in-order per wave; wait for stores to land in LDS.
        asm volatile("s_wait_dscnt 0x0" ::: "memory");

        // ---- A fragments from LDS (ISA 16-bit A layout, ds_load_b128 x2/chunk)
        union AF { v16bf v; v8bf h[2]; };
        const __bf16* fr = &feat[wave][colL][0];
        AF a[4];
#pragma unroll
        for (int c = 0; c < 4; ++c) {
            a[c].h[0] = *(const v8bf*)(fr + 32 * c + hi8);
            a[c].h[1] = *(const v8bf*)(fr + 32 * c + 16 + hi8);
        }
        asm volatile("" ::: "memory");  // keep next iter's DS stores below these loads

        // ---- 16 x v_wmma_f32_16x16x32_bf16 : [16x128] @ [128x64]
        // c-outer / t-inner: consecutive WMMAs hit independent accumulators.
        v8f acc[4] = {{}, {}, {}, {}};
#pragma unroll
        for (int c = 0; c < 4; ++c)
#pragma unroll
            for (int t = 0; t < 4; ++t)
                acc[t] = __builtin_amdgcn_wmma_f32_16x16x32_bf16(
                             false, a[c].v, false, bf[t][c], (short)0, acc[t],
                             false, false);

        // ---- bias + relu + scatter-max (u32 max == f32 max for v >= 0)
        if (base + 16 <= E) {                   // wave-uniform fast path
#pragma unroll
            for (int r = 0; r < 8; ++r) {
                int d0 = __builtin_amdgcn_readlane(myIdx, r);
                int d1 = __builtin_amdgcn_readlane(myIdx, r + 8);
                int d  = half ? d1 : d0;
                unsigned int* po = outU + (size_t)d * C_OUT + colL;
#pragma unroll
                for (int t = 0; t < 4; ++t) {
                    float v = fmaxf(acc[t][r] + bv[t], 0.0f);
                    atomicMax(po + t * 16, __float_as_uint(v));
                }
            }
        } else {                                // tail tile only
#pragma unroll
            for (int r = 0; r < 8; ++r) {
                int d0 = __builtin_amdgcn_readlane(myIdx, r);
                int d1 = __builtin_amdgcn_readlane(myIdx, r + 8);
                int d  = half ? d1 : d0;
                bool ok = (base + hi8 + r) < E;
                unsigned int* po = outU + (size_t)d * C_OUT + colL;
#pragma unroll
                for (int t = 0; t < 4; ++t) {
                    float v = fmaxf(acc[t][r] + bv[t], 0.0f);
                    if (ok) atomicMax(po + t * 16, __float_as_uint(v));
                }
            }
        }
    }
}

// -------------------------------------------------------------------------
extern "C" void kernel_launch(void* const* d_in, const int* in_sizes, int n_in,
                              void* d_out, int out_size, void* d_ws, size_t ws_size,
                              hipStream_t stream) {
    const float* x    = (const float*)d_in[0];
    const int*   eidx = (const int*)  d_in[1];
    const float* W    = (const float*)d_in[2];
    const float* b    = (const float*)d_in[3];
    float*       out  = (float*)d_out;

    const int E = in_sizes[1] / 2;
    const int* srcp = eidx;          // edge_index[0]
    const int* dstp = eidx + E;      // edge_index[1]
    unsigned short* wpk = (unsigned short*)d_ws;   // 16 KB packed bf16 W

    ec_zero<<<(out_size + 255) / 256, 256, 0, stream>>>(out, out_size);
    ec_pack_w<<<2, 256, 0, stream>>>(W, wpk);

    const int nTiles = (E + 15) / 16;
    int blocks = (nTiles + WAVES - 1) / WAVES;
    if (blocks > 1024) blocks = 1024;            // persistent waves
    const int totalWaves = blocks * WAVES;
    const int nIter = (nTiles + totalWaves - 1) / totalWaves;

    ec_edgeconv_wmma<<<blocks, WAVES * 32, 0, stream>>>(
        x, srcp, dstp, wpk, b, out, E, nTiles, nIter);
}
